// MSDeformAttnTransformer_55542517072598
// MI455X (gfx1250) — matrix operand order repaired and checked by reading, not compiled, not run on hardware
//
#include <hip/hip_runtime.h>

typedef __bf16 bf16;
typedef __attribute__((ext_vector_type(16))) __bf16 v16bf;
typedef __attribute__((ext_vector_type(8)))  float  v8f;

#define NB    4
#define NLQ   5440
#define ND    256
#define NH    8
#define NHD   32
#define NLAY  6
#define NFF   1024
#define MROWS (NB * NLQ)   // 21760 = 170 * 128

union FragU { uint4 q[2]; v16bf v; };

// Async DMA: global -> LDS, 16 bytes per lane, tracked by ASYNCcnt.
__device__ __forceinline__ void async_copy_b128(unsigned lds_byte, const void* g) {
  asm volatile("global_load_async_to_lds_b128 %0, %1, off"
               :: "v"(lds_byte), "v"((unsigned long long)(uintptr_t)g)
               : "memory");
}
__device__ __forceinline__ void wait_async0() {
  asm volatile("s_wait_asynccnt 0" ::: "memory");
}

// ---------------------------------------------------------------------------
// Tiled bf16 WMMA GEMM: C[M,N] = A[M,K](bf16 row-major) * Bp (packed WMMA
// fragment order) + bias.
// 256 threads (8 waves); block tile 128x128; K-step 32.
// Wave (waveM 0..3, waveN 0..1) computes 32x64 = 2x4 WMMA fragments.
// A double-buffered in LDS via GLOBAL_LOAD_ASYNC_TO_LDS_B128 (ASYNCcnt);
// B fragments loaded straight from L2-resident packed weights.
// Dynamic LDS: 2 * 128*32 bf16 = 16384 bytes (segment starts at offset 0).
// ---------------------------------------------------------------------------
template <bool RELU, bool STOREF, bool STOREB>
__global__ __launch_bounds__(256) void gemm_bf16_wmma(
    const bf16* __restrict__ A, const bf16* __restrict__ Bp,
    const float* __restrict__ bias,
    float* __restrict__ Cf, bf16* __restrict__ Cb,
    int M, int N, int K)
{
  extern __shared__ bf16 AsDyn[];   // [2][128*32]

  const int tid    = threadIdx.x;
  const int lane   = tid & 31;
  const int wid    = tid >> 5;
  const int waveM  = wid >> 1;
  const int waveN  = wid & 1;
  const int rowBlk = blockIdx.x * 128;
  const int colBlk = blockIdx.y * 128;
  const int nTiles = N >> 4;

  v8f acc[2][4] = {};

  // A staging: thread -> (row, 16-elem segment); two b128 DMAs per thread
  const int aRow = tid >> 1;          // 0..127
  const int aSeg = (tid & 1) * 16;    // 0 / 16
  const unsigned ldsElem = (unsigned)(aRow * 32 + aSeg);
  // A fragment addressing (contiguous k order: k = 16*(lane>>4) + e)
  const int mA  = waveM * 32 + (lane & 15);
  const int kbA = (lane >> 4) * 16;

  const int nsteps = K >> 5;

  auto stageA = [&](int s) {
    const bf16* ag = A + (size_t)(rowBlk + aRow) * K + s * 32 + aSeg;
    const unsigned lb = ((s & 1) ? 4096u * 2u : 0u) + ldsElem * 2u;
    async_copy_b128(lb, ag);
    async_copy_b128(lb + 16u, ag + 8);
  };

  stageA(0);

  for (int s = 0; s < nsteps; ++s) {
    wait_async0();       // my buf-s data landed in LDS
    __syncthreads();     // everyone's landed; all waves done reading buf s^1
    if (s + 1 < nsteps) {
      stageA(s + 1);     // DMA next tile while WMMAs run
      __builtin_prefetch(
          Bp + ((size_t)(s + 1) * nTiles + (colBlk >> 4) + waveN * 4) * 512 +
              lane * 16, 0, 3);
    }

    // B fragments: 32B contiguous per lane from packed weights
    const bf16* bbase =
        Bp + ((size_t)s * nTiles + (colBlk >> 4) + waveN * 4) * 512 + lane * 16;
    v16bf bfr[4];
#pragma unroll
    for (int ni = 0; ni < 4; ++ni)
      bfr[ni] = *(const v16bf*)(bbase + (size_t)ni * 512);

    // A fragments from LDS (2x ds_load_b128 each)
    const bf16* asb = AsDyn + (s & 1) * 4096;
    v16bf afr[2];
#pragma unroll
    for (int mi = 0; mi < 2; ++mi) {
      const bf16* ap = asb + (mA + mi * 16) * 32;
      FragU u;
      u.q[0] = *(const uint4*)(ap + kbA);        // e0..7  : K = kbA..kbA+7
      u.q[1] = *(const uint4*)(ap + kbA + 8);    // e8..15 : K = kbA+8..kbA+15
      afr[mi] = u.v;
    }

#pragma unroll
    for (int mi = 0; mi < 2; ++mi)
#pragma unroll
      for (int ni = 0; ni < 4; ++ni)
        acc[mi][ni] = __builtin_amdgcn_wmma_f32_16x16x32_bf16(
            false, afr[mi], false, bfr[ni], (short)0, acc[mi][ni], false, false);
  }

  // Branch-free epilogue (template bools fold)
  const int cRow = (lane >> 4) * 8;
  const int cCol = lane & 15;
#pragma unroll
  for (int mi = 0; mi < 2; ++mi) {
#pragma unroll
    for (int ni = 0; ni < 4; ++ni) {
      const int colB = colBlk + waveN * 64 + ni * 16 + cCol;
      const float bv = bias[colB];
#pragma unroll
      for (int r = 0; r < 8; ++r) {
        const int row = rowBlk + waveM * 32 + mi * 16 + cRow + r;
        float v = acc[mi][ni][r] + bv;
        if (RELU) v = fmaxf(v, 0.0f);
        if (STOREF) Cf[(size_t)row * N + colB] = v;
        if (STOREB) Cb[(size_t)row * N + colB] = (bf16)v;
      }
    }
  }
}

// ---------------------------------------------------------------------------
// Weight pre-pack: [NLAY,K,N] f32 row-major -> bf16 WMMA-fragment order:
//   n = nt*16 + (lane&15), k = kt*32 + (lane>>4)*16 + e
// ---------------------------------------------------------------------------
__global__ __launch_bounds__(256) void pack_b_kernel(
    const float* __restrict__ W, bf16* __restrict__ dst, int K, int N, int total)
{
  const int i = blockIdx.x * 256 + threadIdx.x;
  if (i >= total) return;
  const int KN  = K * N;
  const int lay = i / KN;
  const int rem = i % KN;
  const int e   = rem & 15;
  const int li  = (rem >> 4) & 31;
  const int t   = rem >> 9;
  const int nTiles = N >> 4;
  const int kt = t / nTiles;
  const int nt = t % nTiles;
  const int n  = nt * 16 + (li & 15);
  const int k  = kt * 32 + (li >> 4) * 16 + e;
  dst[i] = (bf16)W[(size_t)lay * KN + (size_t)k * N + n];
}

// N=384 concat of W_off [NLAY,256,256] and W_attn [NLAY,256,128]
__global__ __launch_bounds__(256) void pack_woa_kernel(
    const float* __restrict__ Woff, const float* __restrict__ Wattn,
    bf16* __restrict__ dst, int total)
{
  const int i = blockIdx.x * 256 + threadIdx.x;
  if (i >= total) return;
  const int KN  = 256 * 384;
  const int lay = i / KN;
  const int rem = i % KN;
  const int e   = rem & 15;
  const int li  = (rem >> 4) & 31;
  const int t   = rem >> 9;
  const int kt  = t / 24;       // nTiles = 24
  const int nt  = t % 24;
  const int n   = nt * 16 + (li & 15);
  const int k   = kt * 32 + (li >> 4) * 16 + e;
  float v;
  if (n < 256) v = Woff[((size_t)lay * 256 + k) * 256 + n];
  else         v = Wattn[((size_t)lay * 256 + k) * 128 + (n - 256)];
  dst[i] = (bf16)v;
}

__global__ __launch_bounds__(256) void concat_boa_kernel(
    const float* __restrict__ boff, const float* __restrict__ battn,
    float* __restrict__ d, int n)
{
  const int i = blockIdx.x * 256 + threadIdx.x;
  if (i >= n) return;
  const int c = i % 384;
  const int lay = i / 384;
  d[i] = (c < 256) ? boff[lay * 256 + c] : battn[lay * 128 + (c - 256)];
}

// ---------------------------------------------------------------------------
// Pack: src/pos [B,D,Hl,Wl] per level -> out [B,LQ,D] (f32 + bf16), pos+emb
// ---------------------------------------------------------------------------
__global__ __launch_bounds__(256) void pack_kernel(
    const float* __restrict__ s0, const float* __restrict__ p0,
    const float* __restrict__ s1, const float* __restrict__ p1,
    const float* __restrict__ s2, const float* __restrict__ p2,
    const float* __restrict__ s3, const float* __restrict__ p3,
    const float* __restrict__ lvlEmb,
    float* __restrict__ outF, bf16* __restrict__ outB, float* __restrict__ posF)
{
  const size_t idx = (size_t)blockIdx.x * 256 + threadIdx.x;  // B*LQ*D
  const int d  = (int)(idx % ND);
  const size_t t = idx / ND;
  const int lq = (int)(t % NLQ);
  const int b  = (int)(t / NLQ);
  int l, s, hw; const float *sp, *pp;
  if      (lq < 4096) { l = 0; s = lq;        sp = s0; pp = p0; hw = 4096; }
  else if (lq < 5120) { l = 1; s = lq - 4096; sp = s1; pp = p1; hw = 1024; }
  else if (lq < 5376) { l = 2; s = lq - 5120; sp = s2; pp = p2; hw = 256;  }
  else                { l = 3; s = lq - 5376; sp = s3; pp = p3; hw = 64;   }
  const float v = sp[((size_t)b * ND + d) * hw + s];
  const float p = pp[((size_t)b * ND + d) * hw + s] + lvlEmb[l * ND + d];
  outF[idx] = v;
  outB[idx] = (bf16)v;
  posF[idx] = p;
}

__global__ __launch_bounds__(256) void q_kernel(
    const float* __restrict__ outF, const float* __restrict__ posF,
    bf16* __restrict__ qB)
{
  const size_t i = (size_t)blockIdx.x * 256 + threadIdx.x;  // exact fit
  qB[i] = (bf16)(outF[i] + posF[i]);
}

// ---------------------------------------------------------------------------
// Softmax over 16 (level,point) logits + deformable bilinear sampling.
// One wave per (b,lq,h); lane = channel (HD == 32 == wave32).
// ---------------------------------------------------------------------------
__global__ __launch_bounds__(256) void sampler_kernel(
    const float* __restrict__ oa, const float* __restrict__ v,
    bf16* __restrict__ samp)
{
  const int gw   = blockIdx.x * 8 + (threadIdx.x >> 5);   // exact fit: 174080
  const int lane = threadIdx.x & 31;
  const int h  = gw % NH;
  const int t  = gw / NH;
  const int lq = t % NLQ;
  const int b  = t / NLQ;

  const int lvlBase[4] = {0, 4096, 5120, 5376};
  const int lvlW[4]    = {64, 32, 16, 8};
  const int lvlH[4]    = {64, 32, 16, 8};

  int lr, s;
  if      (lq < 4096) { lr = 0; s = lq;        }
  else if (lq < 5120) { lr = 1; s = lq - 4096; }
  else if (lq < 5376) { lr = 2; s = lq - 5120; }
  else                { lr = 3; s = lq - 5376; }
  const float rx = ((float)(s % lvlW[lr]) + 0.5f) / (float)lvlW[lr];
  const float ry = ((float)(s / lvlW[lr]) + 0.5f) / (float)lvlH[lr];

  const float* rowOA = oa + (size_t)(b * NLQ + lq) * 384;

  float x = 0.f, y = 0.f, logit = -1e30f;
  if (lane < 16) {
    const int l = lane >> 2;
    const float ox = rowOA[(h * 16 + lane) * 2 + 0];
    const float oy = rowOA[(h * 16 + lane) * 2 + 1];
    logit = rowOA[256 + h * 16 + lane];
    x = (rx + ox / (float)lvlW[l]) * (float)lvlW[l] - 0.5f;
    y = (ry + oy / (float)lvlH[l]) * (float)lvlH[l] - 0.5f;
  }
  float m = logit;
  for (int o = 16; o >= 1; o >>= 1) m = fmaxf(m, __shfl_xor(m, o, 32));
  float e = (lane < 16) ? __expf(logit - m) : 0.f;
  float sum = e;
  for (int o = 16; o >= 1; o >>= 1) sum += __shfl_xor(sum, o, 32);
  const float w = e / sum;

  const float* vb = v + (size_t)b * NLQ * ND;
  float acc = 0.f;
  for (int j = 0; j < 16; ++j) {
    const float xj = __shfl(x, j, 32);
    const float yj = __shfl(y, j, 32);
    const float wj = __shfl(w, j, 32);
    const int l  = j >> 2;
    const int Ws = lvlW[l], Hs = lvlH[l], base = lvlBase[l];
    const float fx = floorf(xj), fy = floorf(yj);
    const int x0 = (int)fx, y0 = (int)fy;
    const float wx = xj - fx, wy = yj - fy;
#pragma unroll
    for (int c = 0; c < 4; ++c) {
      const int xi = x0 + (c & 1);
      const int yi = y0 + (c >> 1);
      const float cw = ((c & 1) ? wx : 1.f - wx) * ((c >> 1) ? wy : 1.f - wy);
      if (xi >= 0 && xi < Ws && yi >= 0 && yi < Hs) {
        acc += wj * cw * vb[(size_t)(base + yi * Ws + xi) * ND + h * NHD + lane];
      }
    }
  }
  samp[(size_t)(b * NLQ + lq) * ND + h * NHD + lane] = (bf16)acc;
}

// ---------------------------------------------------------------------------
// LayerNorm(res + delta)*g + b -> outF (f32) + outB (bf16). Wave per row.
// ---------------------------------------------------------------------------
__global__ __launch_bounds__(256) void ln_kernel(
    const float* __restrict__ res, const float* __restrict__ delta,
    const float* __restrict__ g, const float* __restrict__ bt,
    float* __restrict__ outF, bf16* __restrict__ outB)
{
  const int row  = blockIdx.x * 8 + (threadIdx.x >> 5);   // exact fit
  const int lane = threadIdx.x & 31;
  const float* xr = res   + (size_t)row * ND;
  const float* dr = delta + (size_t)row * ND;
  float vals[8];
  float s = 0.f;
#pragma unroll
  for (int k = 0; k < 8; ++k) {
    vals[k] = xr[k * 32 + lane] + dr[k * 32 + lane];
    s += vals[k];
  }
  for (int o = 16; o >= 1; o >>= 1) s += __shfl_xor(s, o, 32);
  const float mean = s * (1.0f / 256.0f);
  float vs = 0.f;
#pragma unroll
  for (int k = 0; k < 8; ++k) { const float d = vals[k] - mean; vs += d * d; }
  for (int o = 16; o >= 1; o >>= 1) vs += __shfl_xor(vs, o, 32);
  const float inv = rsqrtf(vs * (1.0f / 256.0f) + 1e-5f);
#pragma unroll
  for (int k = 0; k < 8; ++k) {
    const int c = k * 32 + lane;
    const float o = (vals[k] - mean) * inv * g[c] + bt[c];
    outF[(size_t)row * ND + c] = o;
    outB[(size_t)row * ND + c] = (bf16)o;
  }
}

// ---------------------------------------------------------------------------
// Host driver
// ---------------------------------------------------------------------------
extern "C" void kernel_launch(void* const* d_in, const int* in_sizes, int n_in,
                              void* d_out, int out_size, void* d_ws, size_t ws_size,
                              hipStream_t stream) {
  (void)in_sizes; (void)n_in; (void)out_size; (void)ws_size;
  const float* src[4] = {(const float*)d_in[0], (const float*)d_in[2],
                         (const float*)d_in[4], (const float*)d_in[6]};
  const float* pos[4] = {(const float*)d_in[1], (const float*)d_in[3],
                         (const float*)d_in[5], (const float*)d_in[7]};
  const float* level_embed = (const float*)d_in[8];
  const float* W_off  = (const float*)d_in[9];
  const float* b_off  = (const float*)d_in[10];
  const float* W_attn = (const float*)d_in[11];
  const float* b_attn = (const float*)d_in[12];
  const float* W_v    = (const float*)d_in[13];
  const float* b_v    = (const float*)d_in[14];
  const float* W_o    = (const float*)d_in[15];
  const float* b_o    = (const float*)d_in[16];
  const float* ln1_g  = (const float*)d_in[17];
  const float* ln1_b  = (const float*)d_in[18];
  const float* W_fc1  = (const float*)d_in[19];
  const float* b_fc1  = (const float*)d_in[20];
  const float* W_fc2  = (const float*)d_in[21];
  const float* b_fc2  = (const float*)d_in[22];
  const float* ln2_g  = (const float*)d_in[23];
  const float* ln2_b  = (const float*)d_in[24];

  char* ws = (char*)d_ws;
  size_t off = 0;
  auto alloc = [&](size_t bytes) -> char* {
    off = (off + 255) & ~(size_t)255;
    char* p = ws + off;
    off += bytes;
    return p;
  };
  const size_t actF = (size_t)MROWS * ND * sizeof(float);
  const size_t actB = (size_t)MROWS * ND * sizeof(bf16);

  float* outF  = (float*)alloc(actF);
  float* posF  = (float*)alloc(actF);
  bf16*  qB    = (bf16*)alloc(actB);
  bf16*  outB  = (bf16*)alloc(actB);
  float* vF    = (float*)alloc(actF);
  float* oaF   = (float*)alloc((size_t)MROWS * 384 * sizeof(float));
  bf16*  sampB = (bf16*)alloc(actB);
  float* attnF = (float*)alloc(actF);
  float* ln1F  = (float*)alloc(actF);
  bf16*  ln1B  = (bf16*)alloc(actB);
  bf16*  h1B   = (bf16*)alloc((size_t)MROWS * NFF * sizeof(bf16));
  float* ffnF  = (float*)alloc(actF);
  bf16*  WvB   = (bf16*)alloc((size_t)NLAY * ND * ND * sizeof(bf16));
  bf16*  WoaB  = (bf16*)alloc((size_t)NLAY * ND * 384 * sizeof(bf16));
  bf16*  WoB   = (bf16*)alloc((size_t)NLAY * ND * ND * sizeof(bf16));
  bf16*  Wf1B  = (bf16*)alloc((size_t)NLAY * ND * NFF * sizeof(bf16));
  bf16*  Wf2B  = (bf16*)alloc((size_t)NLAY * NFF * ND * sizeof(bf16));
  float* boaF  = (float*)alloc((size_t)NLAY * 384 * sizeof(float));

  auto cdiv = [](int a, int b) { return (a + b - 1) / b; };

  // ---- one-time setup ----
  pack_kernel<<<MROWS, 256, 0, stream>>>(src[0], pos[0], src[1], pos[1],
                                         src[2], pos[2], src[3], pos[3],
                                         level_embed, outF, outB, posF);
  {
    int n = NLAY * ND * ND;
    pack_b_kernel<<<cdiv(n, 256), 256, 0, stream>>>(W_v, WvB, ND, ND, n);
    pack_b_kernel<<<cdiv(n, 256), 256, 0, stream>>>(W_o, WoB, ND, ND, n);
    n = NLAY * ND * NFF;
    pack_b_kernel<<<cdiv(n, 256), 256, 0, stream>>>(W_fc1, Wf1B, ND, NFF, n);
    n = NLAY * NFF * ND;
    pack_b_kernel<<<cdiv(n, 256), 256, 0, stream>>>(W_fc2, Wf2B, NFF, ND, n);
    n = NLAY * ND * 384;
    pack_woa_kernel<<<cdiv(n, 256), 256, 0, stream>>>(W_off, W_attn, WoaB, n);
    n = NLAY * 384;
    concat_boa_kernel<<<cdiv(n, 256), 256, 0, stream>>>(b_off, b_attn, boaF, n);
  }

  const dim3 blk(256);
  const dim3 g256(MROWS / 128, 256 / 128);
  const dim3 g384(MROWS / 128, 384 / 128);
  const dim3 g1024(MROWS / 128, NFF / 128);
  const size_t lds = 2 * 128 * 32 * sizeof(bf16);   // 16384 B dynamic LDS

  for (int i = 0; i < NLAY; ++i) {
    q_kernel<<<MROWS, blk, 0, stream>>>(outF, posF, qB);
    // v = out @ W_v + b_v
    gemm_bf16_wmma<false, true, false><<<g256, blk, lds, stream>>>(
        outB, WvB + (size_t)i * ND * ND, b_v + i * ND,
        vF, nullptr, MROWS, ND, ND);
    // [off | attn_logits] = q @ [W_off|W_attn] + [b_off|b_attn]
    gemm_bf16_wmma<false, true, false><<<g384, blk, lds, stream>>>(
        qB, WoaB + (size_t)i * ND * 384, boaF + i * 384,
        oaF, nullptr, MROWS, 384, ND);
    sampler_kernel<<<MROWS * NH / 8, blk, 0, stream>>>(oaF, vF, sampB);
    // attn = sampled @ W_o + b_o
    gemm_bf16_wmma<false, true, false><<<g256, blk, lds, stream>>>(
        sampB, WoB + (size_t)i * ND * ND, b_o + i * ND,
        attnF, nullptr, MROWS, ND, ND);
    ln_kernel<<<MROWS / 8, blk, 0, stream>>>(
        outF, attnF, ln1_g + i * ND, ln1_b + i * ND, ln1F, ln1B);
    // h1 = relu(out1 @ W_fc1 + b_fc1) -> bf16 only
    gemm_bf16_wmma<true, false, true><<<g1024, blk, lds, stream>>>(
        ln1B, Wf1B + (size_t)i * ND * NFF, b_fc1 + i * NFF,
        nullptr, h1B, MROWS, NFF, ND);
    // ffn = h1 @ W_fc2 + b_fc2
    gemm_bf16_wmma<false, true, false><<<g256, blk, lds, stream>>>(
        h1B, Wf2B + (size_t)i * NFF * ND, b_fc2 + i * ND,
        ffnF, nullptr, MROWS, ND, NFF);
    float* dst = (i == NLAY - 1) ? (float*)d_out : outF;
    ln_kernel<<<MROWS / 8, blk, 0, stream>>>(
        ln1F, ffnF, ln2_g + i * ND, ln2_b + i * ND, dst, outB);
  }
}
